// DiffusionActor_74500502716469
// MI455X (gfx1250) — matrix operand order repaired and checked by reading, not compiled, not run on hardware
//
#include <hip/hip_runtime.h>
#include <hip/hip_bf16.h>

typedef __bf16 bf16_t;
typedef __attribute__((ext_vector_type(16))) __bf16 bf16x16;
typedef __attribute__((ext_vector_type(8)))  __bf16 bf16x8;
typedef __attribute__((ext_vector_type(8)))  float  f32x8;

#define BROWS   64      // batch rows per workgroup
#define THREADS 256     // 8 waves of 32
#define HID     256
#define K1PAD   416     // 386 padded to 13*32
#define IN_DIMC 386
#define ADIM    129
#define NP3     160     // 129 padded to 10*16
#define NSTEPS  30

// ---------------- WMMA GEMM over a tile row -------------------------------
// A: LDS, row-major [rows][a_stride] bf16. B: global, N-major [n][BSTRIDE] bf16.
// Wave computes NT 16x16 D tiles: rows mtbase..mtbase+15, cols nbase + j*16.
template<int NT, int NKT, int BSTRIDE>
__device__ __forceinline__ void gemm_tiles(const bf16_t* __restrict__ Bt,
                                           const bf16_t* A, int a_stride,
                                           int mtbase, int nbase, int lane,
                                           f32x8* acc)
{
    const int l15 = lane & 15;
    const int lh  = lane >> 4;
    const bf16_t* arow  = A  + (mtbase + l15) * a_stride + 16 * lh;
    const bf16_t* bbase = Bt + (size_t)(nbase + l15) * BSTRIDE + 16 * lh;
#pragma unroll 1
    for (int kt = 0; kt < NKT; ++kt) {
        const int kb = kt * 32;
        // A 16x32 bf16: lane l15 = row; VGPR0-3 hold K = 16*lh+0..7, VGPR4-7 hold K = 16*lh+8..15
        bf16x8 alo = *(const bf16x8*)(arow + kb);
        bf16x8 ahi = *(const bf16x8*)(arow + kb + 8);
        bf16x16 a;
#pragma unroll
        for (int i = 0; i < 8; ++i) { a[i] = alo[i]; a[i + 8] = ahi[i]; }
#pragma unroll
        for (int j = 0; j < NT; ++j) {
            // B 32x16 bf16: lane l15 = col n; lanes 0-15 K=0..15, lanes 16-31 K=16..31 (contiguous in N-major W)
            bf16x16 b = *(const bf16x16*)(bbase + (size_t)j * 16 * BSTRIDE + kb);
            acc[j] = __builtin_amdgcn_wmma_f32_16x16x32_bf16(
                false, a, false, b, (short)0, acc[j], false, false);
        }
    }
}

// ---------------- LayerNorm + ReLU in place on LDS bf16 [64][256] ----------
__device__ __forceinline__ void ln_relu_inplace(bf16_t* act, float* part, float* stat,
                                                const float* __restrict__ g,
                                                const float* __restrict__ be, int tid)
{
    const int row = tid >> 2;
    const int q   = tid & 3;
    float s = 0.0f, s2 = 0.0f;
#pragma unroll 4
    for (int c = q * 64; c < q * 64 + 64; ++c) {
        float v = (float)act[row * HID + c];
        s += v; s2 += v * v;
    }
    part[tid * 2]     = s;
    part[tid * 2 + 1] = s2;
    __syncthreads();
    if (q == 0) {
        float S = 0.0f, S2 = 0.0f;
#pragma unroll
        for (int k = 0; k < 4; ++k) {
            S  += part[(row * 4 + k) * 2];
            S2 += part[(row * 4 + k) * 2 + 1];
        }
        float m   = S * (1.0f / 256.0f);
        float var = S2 * (1.0f / 256.0f) - m * m;
        stat[row * 2]     = m;
        stat[row * 2 + 1] = __frsqrt_rn(var + 1e-5f);
    }
    __syncthreads();
    const float m  = stat[row * 2];
    const float rs = stat[row * 2 + 1];
#pragma unroll 4
    for (int c = q * 64; c < q * 64 + 64; ++c) {
        float v = (float)act[row * HID + c];
        float y = (v - m) * rs * g[c] + be[c];
        act[row * HID + c] = (bf16_t)fmaxf(y, 0.0f);
    }
    __syncthreads();
}

// ---------------- weight prep: transpose + pad + bf16 ----------------------
__global__ void prep_weights(const float* __restrict__ W1, const float* __restrict__ W2,
                             const float* __restrict__ W3,
                             bf16_t* __restrict__ W1t, bf16_t* __restrict__ W2t,
                             bf16_t* __restrict__ W3t)
{
    const int T1 = 256 * K1PAD;           // 106496
    const int T2 = 256 * 256;             // 65536
    const int T3 = NP3 * 256;             // 40960
    int i = blockIdx.x * blockDim.x + threadIdx.x;
    if (i < T1) {
        int n = i / K1PAD, k = i % K1PAD;
        W1t[i] = (k < IN_DIMC) ? (bf16_t)W1[k * 256 + n] : (bf16_t)0.0f;
    } else if (i < T1 + T2) {
        int j = i - T1, n = j >> 8, k = j & 255;
        W2t[j] = (bf16_t)W2[k * 256 + n];
    } else if (i < T1 + T2 + T3) {
        int j = i - T1 - T2, n = j >> 8, k = j & 255;
        W3t[j] = (n < ADIM) ? (bf16_t)W3[k * ADIM + n] : (bf16_t)0.0f;
    }
}

// ---------------- main diffusion rollout kernel ----------------------------
__global__ __launch_bounds__(THREADS)
void diffusion_actor_kernel(const float* __restrict__ state,
                            const float* __restrict__ amask,
                            const float* __restrict__ x_init,
                            const float* __restrict__ gumbel,
                            const bf16_t* __restrict__ W1t,
                            const bf16_t* __restrict__ W2t,
                            const bf16_t* __restrict__ W3t,
                            const float* __restrict__ b1, const float* __restrict__ g1,
                            const float* __restrict__ be1,
                            const float* __restrict__ b2, const float* __restrict__ g2,
                            const float* __restrict__ be2,
                            const float* __restrict__ b3,
                            float* __restrict__ out)
{
    __shared__ __align__(32) bf16_t sh_h[BROWS * K1PAD];   // 53248 B  concat input
    __shared__ __align__(32) bf16_t sh_act[BROWS * HID];   // 32768 B  activations
    __shared__ float sh_x[BROWS * ADIM];                   // 33024 B  diffusion state
    __shared__ float sh_pred[BROWS * NP3];                 // 40960 B  noise_pred
    __shared__ float sh_part[BROWS * 4 * 2];               //  2048 B
    __shared__ float sh_stat[BROWS * 2];                   //   512 B

    const int tid    = threadIdx.x;
    const int lane   = tid & 31;
    const int wave   = tid >> 5;
    const int mtbase = (wave & 3) * 16;   // 4 M tiles of 16 rows
    const int nh     = wave >> 1 >> 1;    // 0 or 1: which half of N
    const int row0   = blockIdx.x * BROWS;
    const int l15 = lane & 15, lh = lane >> 4;

    // ---- one-time init: state -> h[:,0:256] (bf16), zero pad, load x_init
    for (int i = tid; i < BROWS * HID; i += THREADS) {
        int r = i >> 8, c = i & 255;
        sh_h[r * K1PAD + c] = (bf16_t)state[(size_t)(row0 + r) * HID + c];
    }
    for (int i = tid; i < BROWS * (K1PAD - IN_DIMC); i += THREADS) {
        int r = i / (K1PAD - IN_DIMC), c = IN_DIMC + i % (K1PAD - IN_DIMC);
        sh_h[r * K1PAD + c] = (bf16_t)0.0f;
    }
    for (int i = tid; i < BROWS * ADIM; i += THREADS)
        sh_x[i] = x_init[(size_t)row0 * ADIM + i];
    __syncthreads();

#pragma unroll 1
    for (int st = 0; st < NSTEPS; ++st) {
        const float tval = (float)(NSTEPS - 1 - st);

        // ---- rebuild dynamic part of h: x columns + time column
        for (int i = tid; i < BROWS * ADIM; i += THREADS) {
            int r = i / ADIM, c = i % ADIM;
            sh_h[r * K1PAD + HID + c] = (bf16_t)sh_x[i];
        }
        if (tid < BROWS) sh_h[tid * K1PAD + (IN_DIMC - 1)] = (bf16_t)tval;
        __syncthreads();

        // ---- layer 1: [64x416] @ [416x256] -> act (bf16, +b1)
        {
            f32x8 acc[8];
#pragma unroll
            for (int j = 0; j < 8; ++j) acc[j] = 0.0f;
            gemm_tiles<8, 13, K1PAD>(W1t, sh_h, K1PAD, mtbase, nh * 128, lane, acc);
            __syncthreads();
#pragma unroll
            for (int j = 0; j < 8; ++j) {
                int col = nh * 128 + j * 16 + l15;
                float bias = b1[col];
#pragma unroll
                for (int r = 0; r < 8; ++r)
                    sh_act[(mtbase + r + 8 * lh) * HID + col] = (bf16_t)(acc[j][r] + bias);
            }
        }
        __syncthreads();
        ln_relu_inplace(sh_act, sh_part, sh_stat, g1, be1, tid);

        // ---- layer 2: [64x256] @ [256x256] -> act in place (+b2)
        {
            f32x8 acc[8];
#pragma unroll
            for (int j = 0; j < 8; ++j) acc[j] = 0.0f;
            gemm_tiles<8, 8, HID>(W2t, sh_act, HID, mtbase, nh * 128, lane, acc);
            __syncthreads();   // all reads of act complete before overwrite
#pragma unroll
            for (int j = 0; j < 8; ++j) {
                int col = nh * 128 + j * 16 + l15;
                float bias = b2[col];
#pragma unroll
                for (int r = 0; r < 8; ++r)
                    sh_act[(mtbase + r + 8 * lh) * HID + col] = (bf16_t)(acc[j][r] + bias);
            }
        }
        __syncthreads();
        ln_relu_inplace(sh_act, sh_part, sh_stat, g2, be2, tid);

        // ---- layer 3: [64x256] @ [256x160pad] -> pred (fp32)
        {
            f32x8 acc[5];
#pragma unroll
            for (int j = 0; j < 5; ++j) acc[j] = 0.0f;
            gemm_tiles<5, 8, HID>(W3t, sh_act, HID, mtbase, nh * 80, lane, acc);
            __syncthreads();
#pragma unroll
            for (int j = 0; j < 5; ++j) {
                int col = nh * 80 + j * 16 + l15;
                if (col < ADIM) {
#pragma unroll
                    for (int r = 0; r < 8; ++r)
                        sh_pred[(mtbase + r + 8 * lh) * NP3 + col] = acc[j][r];
                }
            }
        }
        __syncthreads();

        // ---- Euler update: x -= 0.1 * (pred + b3)
        for (int i = tid; i < BROWS * ADIM; i += THREADS) {
            int r = i / ADIM, c = i % ADIM;
            sh_x[i] -= 0.1f * (sh_pred[r * NP3 + c] + b3[c]);
        }
        __syncthreads();
    }

    // ---- epilogue: straight-through gumbel one-hot + tanh(step)
    if (tid < BROWS) {
        const int r = tid;
        const size_t R = (size_t)(row0 + r);
        float best = -3.402823466e+38f;
        int bi = 0;
        for (int j = 0; j < ADIM - 1; ++j) {
            float lg = sh_x[r * ADIM + j]
                     + (1.0f - amask[R * (ADIM - 1) + j]) * (-1.0e9f)
                     + gumbel[R * (ADIM - 1) + j];
            if (lg > best) { best = lg; bi = j; }
        }
        for (int j = 0; j < ADIM - 1; ++j)
            out[R * ADIM + j] = (j == bi) ? 1.0f : 0.0f;
        out[R * ADIM + (ADIM - 1)] = tanhf(sh_x[r * ADIM + (ADIM - 1)]);
    }
}

// ---------------------------------------------------------------------------
extern "C" void kernel_launch(void* const* d_in, const int* in_sizes, int n_in,
                              void* d_out, int out_size, void* d_ws, size_t ws_size,
                              hipStream_t stream)
{
    const float* state  = (const float*)d_in[0];
    const float* amask  = (const float*)d_in[1];
    const float* x_init = (const float*)d_in[2];
    const float* gumbel = (const float*)d_in[3];
    const float* W1     = (const float*)d_in[4];
    const float* b1     = (const float*)d_in[5];
    const float* g1     = (const float*)d_in[6];
    const float* be1    = (const float*)d_in[7];
    const float* W2     = (const float*)d_in[8];
    const float* b2     = (const float*)d_in[9];
    const float* g2     = (const float*)d_in[10];
    const float* be2    = (const float*)d_in[11];
    const float* W3     = (const float*)d_in[12];
    const float* b3     = (const float*)d_in[13];
    float* out = (float*)d_out;

    bf16_t* W1t = (bf16_t*)d_ws;                 // 256*416 bf16
    bf16_t* W2t = W1t + 256 * K1PAD;             // 256*256 bf16
    bf16_t* W3t = W2t + 256 * 256;               // 160*256 bf16

    const int B = in_sizes[0] / HID;             // 65536

    const int prep_elems = 256 * K1PAD + 256 * 256 + NP3 * 256;
    prep_weights<<<(prep_elems + 255) / 256, 256, 0, stream>>>(W1, W2, W3, W1t, W2t, W3t);

    diffusion_actor_kernel<<<B / BROWS, THREADS, 0, stream>>>(
        state, amask, x_init, gumbel, W1t, W2t, W3t,
        b1, g1, be1, b2, g2, be2, b3, out);
}
